// SA_66657892434304
// MI455X (gfx1250) — matrix operand order repaired and checked by reading, not compiled, not run on hardware
//
#include <hip/hip_runtime.h>
#include <math.h>

typedef __attribute__((ext_vector_type(16))) __bf16 v16bf;
typedef __attribute__((ext_vector_type(8)))  float  v8f;
typedef __bf16 bf16_t;

// ---------------------------------------------------------------------------
// Elementwise helpers
// ---------------------------------------------------------------------------
__global__ __launch_bounds__(256)
void f2bf_kernel(const float* __restrict__ a, bf16_t* __restrict__ o, long n) {
  long i = (long)blockIdx.x * blockDim.x + threadIdx.x;
  if (i < n) o[i] = (bf16_t)a[i];
}

__global__ __launch_bounds__(256)
void init_slots_kernel(const float* __restrict__ si, float* __restrict__ sf,
                       bf16_t* __restrict__ sb, int per, int total) {
  int i = blockIdx.x * blockDim.x + threadIdx.x;
  if (i >= total) return;
  float v = si[i % per];
  sf[i] = v;
  sb[i] = (bf16_t)v;
}

// Row LayerNorm (dim=768), f32 in -> bf16 out. One block per row.
__global__ __launch_bounds__(256)
void ln_bf16_kernel(const float* __restrict__ x, const float* __restrict__ g,
                    const float* __restrict__ be, bf16_t* __restrict__ out, int dim) {
  int row = blockIdx.x;
  const float* xr = x + (size_t)row * dim;
  float s = 0.f, s2 = 0.f;
  for (int i = threadIdx.x; i < dim; i += blockDim.x) { float v = xr[i]; s += v; s2 += v * v; }
  for (int off = 16; off > 0; off >>= 1) {
    s  += __shfl_down(s,  off, 32);
    s2 += __shfl_down(s2, off, 32);
  }
  __shared__ float ps[8], ps2[8], stat[2];
  int wid = threadIdx.x >> 5, lane = threadIdx.x & 31;
  if (lane == 0) { ps[wid] = s; ps2[wid] = s2; }
  __syncthreads();
  if (threadIdx.x == 0) {
    float t1 = 0.f, t2 = 0.f;
    int nw = blockDim.x >> 5;
    for (int i = 0; i < nw; ++i) { t1 += ps[i]; t2 += ps2[i]; }
    float mu  = t1 / dim;
    float var = t2 / dim - mu * mu;
    stat[0] = mu; stat[1] = rsqrtf(var + 1e-5f);
  }
  __syncthreads();
  float mu = stat[0], rs = stat[1];
  bf16_t* orow = out + (size_t)row * dim;
  for (int i = threadIdx.x; i < dim; i += blockDim.x)
    orow[i] = (bf16_t)((xr[i] - mu) * rs * g[i] + be[i]);
}

// GRU combine: r,z,n gates from gi/gh (f32, bias already added by GEMM epilogue)
__global__ __launch_bounds__(256)
void gru_kernel(const float* __restrict__ gi, const float* __restrict__ gh,
                float* __restrict__ slots, bf16_t* __restrict__ slots_bf,
                int rows, int dim) {
  long i = (long)blockIdx.x * blockDim.x + threadIdx.x;
  if (i >= (long)rows * dim) return;
  int r = (int)(i / dim), c = (int)(i % dim);
  const float* gir = gi + (size_t)r * 3 * dim;
  const float* ghr = gh + (size_t)r * 3 * dim;
  float rr = 1.f / (1.f + __expf(-(gir[c] + ghr[c])));
  float z  = 1.f / (1.f + __expf(-(gir[dim + c] + ghr[dim + c])));
  float n  = tanhf(gir[2 * dim + c] + rr * ghr[2 * dim + c]);
  float h  = (1.f - z) * n + z * slots[i];
  slots[i] = h;
  slots_bf[i] = (bf16_t)h;
}

// ---------------------------------------------------------------------------
// Generic WMMA GEMM: C[M,N] = A[M,K](bf16) @ op(B)(bf16), f32 accumulate.
// BT=true : B stored [N,K] (compute A @ B^T, weight-style), 128-bit loads.
// BT=false: B stored [K,N] (attn @ v case).
// Epilogue: optional bias[n], v /= rowdiv[row], relu, + resid, f32/bf16 stores.
// One wave per 16x16 output tile; M,N multiples of 16; K multiple of 32.
// ---------------------------------------------------------------------------
template <bool BT>
__global__ __launch_bounds__(256)
void wmma_gemm_kernel(const bf16_t* __restrict__ A, const bf16_t* __restrict__ Bm,
                      const float* __restrict__ bias, const float* __restrict__ rowdiv,
                      const float* __restrict__ resid, float* __restrict__ Cf,
                      bf16_t* __restrict__ Cb,
                      int M, int N, int K, int lda, int ldb, int ldc,
                      long aB, long bB, long cB, int rdB, float alpha, int relu) {
  int b = blockIdx.y;
  A  += (size_t)b * aB;
  Bm += (size_t)b * bB;
  int wave = threadIdx.x >> 5, lane = threadIdx.x & 31;
  int tile = blockIdx.x * (blockDim.x >> 5) + wave;
  int ntN = N >> 4;
  int totalTiles = (M >> 4) * ntN;
  if (tile >= totalTiles) return;
  int tm = tile / ntN, tn = tile - tm * ntN;
  int row0 = tm << 4, col0 = tn << 4;
  int m = lane & 15;
  bool hi = (lane & 16) != 0;

  v8f acc = {};
  const bf16_t* arow = A + (size_t)(row0 + m) * lda + (hi ? 16 : 0);
  for (int k0 = 0; k0 < K; k0 += 32) {
    union { v16bf v; uint4 u[2]; } af, bfu;
    // A 16x32 bf16 fragment: lane holds row m; 16 contiguous K values,
    // split 16/16 by lane half (matches B).
    const uint4* ap = (const uint4*)(arow + k0);
    af.u[0] = ap[0];
    af.u[1] = ap[1];
    __builtin_prefetch((const void*)(arow + k0 + 128), 0, 2);
    if (BT) {
      // B^T column n = lane m: 32 contiguous K values, split 16/16 by lane half.
      const bf16_t* bp = Bm + (size_t)(col0 + m) * ldb + k0 + (hi ? 16 : 0);
      const uint4* bp4 = (const uint4*)bp;
      bfu.u[0] = bp4[0];
      bfu.u[1] = bp4[1];
      __builtin_prefetch((const void*)(bp + 128), 0, 2);
    } else {
      const bf16_t* bp = Bm + (size_t)(k0 + (hi ? 16 : 0)) * ldb + col0 + m;
#pragma unroll
      for (int h = 0; h < 16; ++h) bfu.v[h] = bp[(size_t)h * ldb];
    }
    acc = __builtin_amdgcn_wmma_f32_16x16x32_bf16(false, af.v, false, bfu.v,
                                                  (short)0, acc, false, false);
  }

  int col = col0 + m;
#pragma unroll
  for (int j = 0; j < 8; ++j) {
    int row = row0 + j + (hi ? 8 : 0);
    float v = acc[j] * alpha;
    if (bias)   v += bias[col];
    if (rowdiv) v /= rowdiv[b * rdB + row];
    if (relu)   v = fmaxf(v, 0.f);
    size_t ci = (size_t)b * cB + (size_t)row * ldc + col;
    if (resid)  v += resid[ci];
    if (Cf) Cf[ci] = v;
    if (Cb) Cb[ci] = (bf16_t)v;
  }
}

// ---------------------------------------------------------------------------
// dots = scale * q @ k^T, masked, softmax over the 16 slots (fully inside one
// wave tile: rows 0-7 in VGPRs, rows 8-15 via shfl_xor(16)), store attn+EPS
// (bf16, A-operand of the updates GEMM) and accumulate per-slot renorm sums.
// ---------------------------------------------------------------------------
__global__ __launch_bounds__(256)
void dots_softmax_kernel(const bf16_t* __restrict__ q,    // [B,16,K]
                         const bf16_t* __restrict__ kbf,  // [B,T,K]
                         const unsigned char* __restrict__ cmask, // [B,T]
                         bf16_t* __restrict__ attn_un,    // [B,16,T]
                         float* __restrict__ attn_vis,    // [B,16,T] or null
                         float* __restrict__ rsum,        // [B,16]
                         int T, int K, float scale) {
  int b = blockIdx.y;
  int wave = threadIdx.x >> 5, lane = threadIdx.x & 31;
  int tile = blockIdx.x * (blockDim.x >> 5) + wave;
  int ntiles = T >> 4;
  if (tile >= ntiles) return;
  int t0 = tile << 4;
  int m = lane & 15;
  bool hi = (lane & 16) != 0;

  const bf16_t* A  = q   + (size_t)b * 16 * K;
  const bf16_t* Bm = kbf + (size_t)b * T * K;
  v8f acc = {};
  const bf16_t* arow = A + (size_t)m * K + (hi ? 16 : 0);
  const bf16_t* brow = Bm + (size_t)(t0 + m) * K + (hi ? 16 : 0);
  for (int k0 = 0; k0 < K; k0 += 32) {
    union { v16bf v; uint4 u[2]; } af, bfu;
    const uint4* ap = (const uint4*)(arow + k0);
    af.u[0] = ap[0];
    af.u[1] = ap[1];
    const uint4* bp = (const uint4*)(brow + k0);
    bfu.u[0] = bp[0];
    bfu.u[1] = bp[1];
    __builtin_prefetch((const void*)(brow + k0 + 128), 0, 2);
    acc = __builtin_amdgcn_wmma_f32_16x16x32_bf16(false, af.v, false, bfu.v,
                                                  (short)0, acc, false, false);
  }

  int t = t0 + m;
  bool ok = cmask[(size_t)b * T + t] != 0;
  const float NEGF = -3.402823466e38f;
  float dv[8];
  float mx = NEGF;
#pragma unroll
  for (int j = 0; j < 8; ++j) { dv[j] = ok ? acc[j] * scale : NEGF; mx = fmaxf(mx, dv[j]); }
  mx = fmaxf(mx, __shfl_xor(mx, 16, 32));   // combine rows 0-7 with rows 8-15
  float e[8], s = 0.f;
#pragma unroll
  for (int j = 0; j < 8; ++j) { e[j] = __expf(dv[j] - mx); s += e[j]; }
  s += __shfl_xor(s, 16, 32);
  float inv = 1.f / s;
  float p[8];
#pragma unroll
  for (int j = 0; j < 8; ++j) {
    float av = e[j] * inv;                       // attn_vis (softmax over slots)
    int i = j + (hi ? 8 : 0);
    size_t idx = ((size_t)b * 16 + i) * T + t;
    if (attn_vis) attn_vis[idx] = av;
    float au = av + 1e-8f;                       // + EPS before T-renorm
    attn_un[idx] = (bf16_t)au;
    p[j] = au;
  }
  // sum over the 16 tokens of this tile (across the 16-lane group), then atomic
#pragma unroll
  for (int off = 1; off < 16; off <<= 1)
#pragma unroll
    for (int j = 0; j < 8; ++j) p[j] += __shfl_xor(p[j], off, 32);
  if (m == 0)
#pragma unroll
    for (int j = 0; j < 8; ++j)
      atomicAdd(&rsum[b * 16 + j + (hi ? 8 : 0)], p[j]);
}

// ---------------------------------------------------------------------------
// cluster_single: 16x16 cosine-distance graph, transitive closure by 4
// squarings, component means + merged attention. One block per batch.
// ---------------------------------------------------------------------------
__global__ __launch_bounds__(256)
void cluster_kernel(const float* __restrict__ slots,  // [B,16,D]
                    const float* __restrict__ attn,   // [B,16,T]
                    float* __restrict__ out_slots,    // [B,16,D]
                    float* __restrict__ out_attn,     // [B,T,16] (transposed)
                    float* __restrict__ out_mask,     // [B,16]
                    int T, int D) {
  int b = blockIdx.x;
  const float* S = slots + (size_t)b * 16 * D;
  __shared__ float dot[16][16], adj[16][16], tmp[16][16], cm[16][16], cw[16][16];
  __shared__ float nrm[16];
  __shared__ int counts_sh;
  int n = threadIdx.x >> 4, mc = threadIdx.x & 15;

  float d = 0.f;
  for (int c = 0; c < D; ++c) d += S[(size_t)n * D + c] * S[(size_t)mc * D + c];
  dot[n][mc] = d;
  __syncthreads();
  if (threadIdx.x < 16) nrm[threadIdx.x] = fmaxf(sqrtf(dot[threadIdx.x][threadIdx.x]), 1e-12f);
  __syncthreads();
  float dist = 1.f - dot[n][mc] / (nrm[n] * nrm[mc]);
  adj[n][mc] = (dist < 0.5f) ? 1.f : 0.f;
  __syncthreads();
  for (int it = 0; it < 4; ++it) {   // ceil(log2(16)) squarings
    float a = 0.f;
    for (int j = 0; j < 16; ++j) a += adj[n][j] * adj[j][mc];
    tmp[n][mc] = fminf(a, 1.f);
    __syncthreads();
    adj[n][mc] = tmp[n][mc];
    __syncthreads();
  }
  float cum = 0.f;
  for (int j = 0; j <= n; ++j) cum += adj[j][mc];
  cm[n][mc] = adj[n][mc] * ((cum <= 1.f) ? 1.f : 0.f);
  __syncthreads();
  if (threadIdx.x == 0) {
    int cnt = 0;
    for (int i = 0; i < 16; ++i) {
      bool any = false;
      for (int j = 0; j < 16; ++j) any |= cm[i][j] > 0.f;
      cnt += any ? 1 : 0;
    }
    counts_sh = cnt;
  }
  __syncthreads();
  float cmval = (counts_sh < 3) ? ((n == mc) ? 1.f : 0.f) : cm[n][mc];
  __syncthreads();
  cm[n][mc] = cmval;
  __syncthreads();
  float rs = 0.f;
  for (int j = 0; j < 16; ++j) rs += cm[n][j];
  cw[n][mc] = cm[n][mc] / fmaxf(rs, 1.f);
  __syncthreads();
  if (threadIdx.x < 16) {
    bool any = false;
    for (int j = 0; j < 16; ++j) any |= cm[threadIdx.x][j] > 0.f;
    out_mask[b * 16 + threadIdx.x] = any ? 1.f : 0.f;
  }
  // means = cw @ slots
  for (int idx = threadIdx.x; idx < 16 * D; idx += blockDim.x) {
    int i = idx / D, c = idx % D;
    float v = 0.f;
    for (int j = 0; j < 16; ++j) v += cw[i][j] * S[(size_t)j * D + c];
    out_slots[(size_t)b * 16 * D + idx] = v;
  }
  // attn_map[b,t,i] = (cm @ attn)[i,t]
  const float* Av = attn + (size_t)b * 16 * T;
  for (int idx = threadIdx.x; idx < T * 16; idx += blockDim.x) {
    int t = idx >> 4, i = idx & 15;
    float v = 0.f;
    for (int j = 0; j < 16; ++j) v += cm[i][j] * Av[(size_t)j * T + t];
    out_attn[(size_t)b * T * 16 + idx] = v;
  }
}

// ---------------------------------------------------------------------------
// Host orchestration
// ---------------------------------------------------------------------------
extern "C" void kernel_launch(void* const* d_in, const int* in_sizes, int n_in,
                              void* d_out, int out_size, void* d_ws, size_t ws_size,
                              hipStream_t stream) {
  (void)in_sizes; (void)n_in; (void)out_size; (void)ws_size;
  const int Bb = 16, T = 4096, D = 768, Sd = 768, NS = 16, NIT = 9;
  const float scale = 1.0f / sqrtf((float)D);

  const float* x        = (const float*)d_in[0];
  const unsigned char* cmask = (const unsigned char*)d_in[1];
  const float* slots_in = (const float*)d_in[3];
  const float* Wk  = (const float*)d_in[4];
  const float* Wv  = (const float*)d_in[5];
  const float* Wq  = (const float*)d_in[6];
  const float* gwih = (const float*)d_in[7];
  const float* gwhh = (const float*)d_in[8];
  const float* gbih = (const float*)d_in[9];
  const float* gbhh = (const float*)d_in[10];
  const float* w1  = (const float*)d_in[11];
  const float* b1  = (const float*)d_in[12];
  const float* w2  = (const float*)d_in[13];
  const float* b2  = (const float*)d_in[14];
  const float* lin_g = (const float*)d_in[15];
  const float* lin_b = (const float*)d_in[16];
  const float* lsg = (const float*)d_in[17];
  const float* lsb = (const float*)d_in[18];
  const float* lfg = (const float*)d_in[19];
  const float* lfb = (const float*)d_in[20];

  // ---- workspace carve ----
  char* wp = (char*)d_ws;
  auto carve = [&](size_t bytes) -> char* {
    char* p = wp;
    wp += (bytes + 255) & ~(size_t)255;
    return p;
  };
  bf16_t* xn   = (bf16_t*)carve((size_t)Bb * T * D * 2);
  bf16_t* kbf  = (bf16_t*)carve((size_t)Bb * T * Sd * 2);
  bf16_t* vbf  = (bf16_t*)carve((size_t)Bb * T * Sd * 2);
  bf16_t* Wk_b = (bf16_t*)carve((size_t)Sd * D * 2);
  bf16_t* Wv_b = (bf16_t*)carve((size_t)Sd * D * 2);
  bf16_t* Wq_b = (bf16_t*)carve((size_t)Sd * Sd * 2);
  bf16_t* gwih_b = (bf16_t*)carve((size_t)3 * Sd * Sd * 2);
  bf16_t* gwhh_b = (bf16_t*)carve((size_t)3 * Sd * Sd * 2);
  bf16_t* w1_b = (bf16_t*)carve((size_t)4 * Sd * Sd * 2);
  bf16_t* w2_b = (bf16_t*)carve((size_t)Sd * 4 * Sd * 2);
  float*  slots_f = (float*)carve((size_t)Bb * NS * Sd * 4);
  bf16_t* slots_b = (bf16_t*)carve((size_t)Bb * NS * Sd * 2);
  bf16_t* qn   = (bf16_t*)carve((size_t)Bb * NS * Sd * 2);   // also MLP "pre"
  bf16_t* qb   = (bf16_t*)carve((size_t)Bb * NS * Sd * 2);
  bf16_t* updb = (bf16_t*)carve((size_t)Bb * NS * Sd * 2);
  bf16_t* h1b  = (bf16_t*)carve((size_t)Bb * NS * 4 * Sd * 2);
  float*  gi   = (float*)carve((size_t)Bb * NS * 3 * Sd * 4);
  float*  gh   = (float*)carve((size_t)Bb * NS * 3 * Sd * 4);
  bf16_t* attn_un  = (bf16_t*)carve((size_t)Bb * NS * T * 2);
  float*  attn_vis = (float*)carve((size_t)Bb * NS * T * 4);
  float*  rsum = (float*)carve((size_t)Bb * NS * 4);

  float* out_slots = (float*)d_out;
  float* out_attn  = out_slots + (size_t)Bb * NS * Sd;
  float* out_mask  = out_attn + (size_t)Bb * T * NS;

  auto gemm = [&](bool bt, const bf16_t* A, const bf16_t* Bm, const float* bias,
                  const float* rowdiv, const float* resid, float* Cf, bf16_t* Cb,
                  int M, int N, int K, int lda, int ldb, int ldc,
                  long aB, long bB, long cB, int rdB, int relu, int batch) {
    int tiles = (M / 16) * (N / 16);
    dim3 grid((tiles + 7) / 8, batch);
    if (bt)
      wmma_gemm_kernel<true><<<grid, 256, 0, stream>>>(A, Bm, bias, rowdiv, resid, Cf, Cb,
          M, N, K, lda, ldb, ldc, aB, bB, cB, rdB, 1.0f, relu);
    else
      wmma_gemm_kernel<false><<<grid, 256, 0, stream>>>(A, Bm, bias, rowdiv, resid, Cf, Cb,
          M, N, K, lda, ldb, ldc, aB, bB, cB, rdB, 1.0f, relu);
  };
  auto conv = [&](const float* src, bf16_t* dst, long n) {
    f2bf_kernel<<<(int)((n + 255) / 256), 256, 0, stream>>>(src, dst, n);
  };

  // --- preprocessing: LN(x) -> bf16, weights -> bf16, k/v projections ---
  ln_bf16_kernel<<<Bb * T, 256, 0, stream>>>(x, lin_g, lin_b, xn, D);
  conv(Wk, Wk_b, (long)Sd * D);
  conv(Wv, Wv_b, (long)Sd * D);
  conv(Wq, Wq_b, (long)Sd * Sd);
  conv(gwih, gwih_b, (long)3 * Sd * Sd);
  conv(gwhh, gwhh_b, (long)3 * Sd * Sd);
  conv(w1, w1_b, (long)4 * Sd * Sd);
  conv(w2, w2_b, (long)Sd * 4 * Sd);
  gemm(true, xn, Wk_b, nullptr, nullptr, nullptr, nullptr, kbf,
       Bb * T, Sd, D, D, D, Sd, 0, 0, 0, 0, 0, 1);
  gemm(true, xn, Wv_b, nullptr, nullptr, nullptr, nullptr, vbf,
       Bb * T, Sd, D, D, D, Sd, 0, 0, 0, 0, 0, 1);
  init_slots_kernel<<<(Bb * NS * Sd + 255) / 256, 256, 0, stream>>>(
      slots_in, slots_f, slots_b, NS * Sd, Bb * NS * Sd);

  // --- 8 warm iterations + 1 final (identical math; masks are all-true) ---
  for (int it = 0; it < NIT; ++it) {
    bool last = (it == NIT - 1);
    // q = LN(slots) @ Wq^T
    ln_bf16_kernel<<<Bb * NS, 256, 0, stream>>>(slots_f, lsg, lsb, qn, Sd);
    gemm(true, qn, Wq_b, nullptr, nullptr, nullptr, nullptr, qb,
         Bb * NS, Sd, Sd, Sd, Sd, Sd, 0, 0, 0, 0, 0, 1);
    // dots + slot-softmax + EPS renorm sums
    hipMemsetAsync(rsum, 0, (size_t)Bb * NS * sizeof(float), stream);
    dots_softmax_kernel<<<dim3(T / 16 / 8, Bb), 256, 0, stream>>>(
        qb, kbf, cmask, attn_un, last ? attn_vis : nullptr, rsum, T, Sd, scale);
    // updates = (attn_un @ v) / rsum
    gemm(false, attn_un, vbf, nullptr, rsum, nullptr, nullptr, updb,
         NS, Sd, T, T, Sd, Sd, (long)NS * T, (long)T * Sd, (long)NS * Sd, NS, 0, Bb);
    // GRU gates
    gemm(true, updb, gwih_b, gbih, nullptr, nullptr, gi, nullptr,
         Bb * NS, 3 * Sd, Sd, Sd, Sd, 3 * Sd, 0, 0, 0, 0, 0, 1);
    gemm(true, slots_b, gwhh_b, gbhh, nullptr, nullptr, gh, nullptr,
         Bb * NS, 3 * Sd, Sd, Sd, Sd, 3 * Sd, 0, 0, 0, 0, 0, 1);
    gru_kernel<<<(Bb * NS * Sd + 255) / 256, 256, 0, stream>>>(
        gi, gh, slots_f, slots_b, Bb * NS, Sd);
    // MLP residual: slots += relu(LN(slots)@w1^T+b1)@w2^T+b2
    ln_bf16_kernel<<<Bb * NS, 256, 0, stream>>>(slots_f, lfg, lfb, qn, Sd);
    gemm(true, qn, w1_b, b1, nullptr, nullptr, nullptr, h1b,
         Bb * NS, 4 * Sd, Sd, Sd, Sd, 4 * Sd, 0, 0, 0, 0, 1, 1);
    gemm(true, h1b, w2_b, b2, nullptr, slots_f, slots_f, slots_b,
         Bb * NS, Sd, 4 * Sd, 4 * Sd, 4 * Sd, Sd, 0, 0, 0, 0, 0, 1);
  }

  // --- clustering + outputs ---
  cluster_kernel<<<Bb, 256, 0, stream>>>(slots_f, attn_vis,
                                         out_slots, out_attn, out_mask, T, Sd);
}
